// Align_MoE_9732395892816
// MI455X (gfx1250) — matrix operand order repaired and checked by reading, not compiled, not run on hardware
//
#include <hip/hip_runtime.h>
#include <hip/hip_bf16.h>
#include <math.h>

// ---------------------------------------------------------------------------
// Problem constants (from reference): B=8, S=512, H=1024, E=8, TOPK=2, D=2H
// N tokens = B*S = 4096, D = 2048
// ---------------------------------------------------------------------------
#define NTOK   4096
#define DDIM   2048
#define HDIM   1024
#define NEXP   8

typedef __attribute__((ext_vector_type(16))) __bf16        v16bf;
typedef __attribute__((ext_vector_type(8)))  float         v8f;
typedef __attribute__((ext_vector_type(4)))  unsigned int  u32x4;

struct FragBits { u32x4 lo, hi; };

static __device__ __forceinline__ v16bf make_frag(u32x4 lo, u32x4 hi) {
    FragBits fb{lo, hi};
    return __builtin_bit_cast(v16bf, fb);
}

static __device__ __forceinline__ unsigned short f2bf(float f) {
    unsigned int x = __float_as_uint(f);
    unsigned int r = x + 0x7FFFu + ((x >> 16) & 1u);   // round-to-nearest-even
    return (unsigned short)(r >> 16);
}

static __device__ __forceinline__ unsigned int pack2bf(float a, float b) {
    return (unsigned int)f2bf(a) | ((unsigned int)f2bf(b) << 16);
}

// ---------------------------------------------------------------------------
// Gate kernel: one block per token; wave w handles expert w for both gates.
// ---------------------------------------------------------------------------
__global__ void gate_kernel(const float* __restrict__ x,
                            const float* __restrict__ Wg, const float* __restrict__ bg,
                            const float* __restrict__ Wf, const float* __restrict__ bfv,
                            float* __restrict__ route_g, float* __restrict__ rout_f) {
    const int n    = blockIdx.x;
    const int wave = threadIdx.x >> 5;
    const int lane = threadIdx.x & 31;
    __shared__ float lg[NEXP], lf[NEXP];

    const float* xr = x + (size_t)n * DDIM;
    float pg = 0.f, pf = 0.f;
    for (int i = lane; i < HDIM; i += 32) {
        pg += xr[i]        * Wg[i * NEXP + wave];
        pf += xr[HDIM + i] * Wf[i * NEXP + wave];
    }
    for (int off = 16; off > 0; off >>= 1) {
        pg += __shfl_down(pg, off, 32);
        pf += __shfl_down(pf, off, 32);
    }
    if (lane == 0) { lg[wave] = pg + bg[wave]; lf[wave] = pf + bfv[wave]; }
    __syncthreads();

    if (threadIdx.x == 0) {
        float mg = -1e30f, mf = -1e30f;
        for (int e = 0; e < NEXP; ++e) { mg = fmaxf(mg, lg[e]); mf = fmaxf(mf, lf[e]); }
        float eg[NEXP], ef[NEXP], sg = 0.f, sf = 0.f;
        for (int e = 0; e < NEXP; ++e) {
            eg[e] = __expf(lg[e] - mg); sg += eg[e];
            ef[e] = __expf(lf[e] - mf); sf += ef[e];
        }
        const float rg_ = 1.f / sg, rf_ = 1.f / sf;
        for (int e = 0; e < NEXP; ++e) { eg[e] *= rg_; ef[e] *= rf_; }
        int i1 = 0;
        for (int e = 1; e < NEXP; ++e) if (eg[e] > eg[i1]) i1 = e;
        int i2 = (i1 == 0) ? 1 : 0;
        for (int e = 0; e < NEXP; ++e) if (e != i1 && eg[e] > eg[i2]) i2 = e;
        for (int e = 0; e < NEXP; ++e) {
            route_g[(size_t)n * NEXP + e] = (e == i1 || e == i2) ? eg[e] : 0.f;
            rout_f [(size_t)n * NEXP + e] = ef[e];
        }
    }
}

// ---------------------------------------------------------------------------
// X (f32) -> bf16 workspace copy.  4 elements per thread.
// ---------------------------------------------------------------------------
__global__ void cvt_bf16_kernel(const float* __restrict__ src,
                                unsigned short* __restrict__ dst, int n4) {
    int i = blockIdx.x * blockDim.x + threadIdx.x;
    if (i >= n4) return;
    float4 v = *(const float4*)(src + (size_t)i * 4);
    unsigned int* d = (unsigned int*)(dst + (size_t)i * 4);
    d[0] = pack2bf(v.x, v.y);
    d[1] = pack2bf(v.z, v.w);
}

// ---------------------------------------------------------------------------
// Zero d_out (harness poisons it; re-init every launch for determinism).
// ---------------------------------------------------------------------------
__global__ void zero_kernel(float* __restrict__ p, int n4) {
    int i = blockIdx.x * blockDim.x + threadIdx.x;
    if (i >= n4) return;
    *(float4*)(p + (size_t)i * 4) = make_float4(0.f, 0.f, 0.f, 0.f);
}

// ---------------------------------------------------------------------------
// BF16 WMMA GEMM, double-buffered LDS software pipeline.
//   C[M,N] = A[M,K] @ W[K,N] (+bias, epilogue)
//   A: bf16 row-major in workspace (X or hidden)
//   W: f32 global expert slice, converted f32->bf16 while staging to LDS
//   mode 1: hidden = relu(C + b1)  -> bf16 Hout
//   mode 2: gated accumulation into d_out (out0 | out1 column split)
// Block: 256 threads (8 waves, 2x4).  Block tile 128x256, K-step 32.
// Wave tile 64x64 = 4x4 fragments -> 16 v_wmma_f32_16x16x32_bf16 per K iter.
// ---------------------------------------------------------------------------
#define BT_M   128
#define BT_N   256
#define KSTEP  32
#define KT     (DDIM / KSTEP)          // 64 iterations
#define LDSTR  40                      // padded stride (bf16 elems), 16B-aligned rows

__global__ void __launch_bounds__(256)
moe_gemm_kernel(const unsigned short* __restrict__ A,     // [NTOK, DDIM] bf16
                const float* __restrict__ Bw,             // [DDIM, DDIM] f32
                const float* __restrict__ bias,           // [DDIM]
                unsigned short* __restrict__ Hout,        // mode 1 output (bf16)
                float* __restrict__ Oout,                 // mode 2 output (d_out)
                const float* __restrict__ route_g,        // [NTOK, NEXP]
                const float* __restrict__ rout_f,         // [NTOK, NEXP]
                const float* __restrict__ wcomb,          // weight[2]
                int expert, int mode) {
    __shared__ __align__(16) unsigned short ldsA[2][BT_M * LDSTR]; // [row][k]
    __shared__ __align__(16) unsigned short ldsB[2][BT_N * LDSTR]; // [col][k] transposed

    const int tid   = threadIdx.x;
    const int wave  = tid >> 5;
    const int lane  = tid & 31;
    const int waveM = wave & 1;             // 0..1  (64 rows each)
    const int waveN = wave >> 1;            // 0..3  (64 cols each)
    const int gm    = blockIdx.y * BT_M;    // 32 M tiles
    const int gn    = blockIdx.x * BT_N;    // 8  N tiles
    const int ml    = lane & 15;
    const int khalf = lane >> 4;

    // ---- staging decomposition (per thread, fixed across iterations) ----
    // A tile: 128 rows x 32 bf16 = 512 16B chunks; 2 chunks per thread.
    const int aRow0 = tid >> 2;             // chunk tid
    const int aSeg0 = tid & 3;
    const int aRow1 = (tid + 256) >> 2;     // chunk tid+256
    const int aSeg1 = (tid + 256) & 3;
    // B tile: 16 K-pairs x 64 N-quads = 1024 items; 4 items per thread.
    // item q: kp = q>>6 (K rows 2kp,2kp+1), n4 = (q&63)*4.

    u32x4  aReg[2];
    float4 bReg[4][2];

    v8f acc[4][4] = {};

    auto load_tiles = [&](int kt) {
        const int k0 = kt * KSTEP;
        aReg[0] = *(const u32x4*)(A + (size_t)(gm + aRow0) * DDIM + k0 + aSeg0 * 8);
        aReg[1] = *(const u32x4*)(A + (size_t)(gm + aRow1) * DDIM + k0 + aSeg1 * 8);
        #pragma unroll
        for (int it = 0; it < 4; ++it) {
            int q  = tid + it * 256;
            int kp = q >> 6;
            int n4 = (q & 63) * 4;
            const float* src = Bw + (size_t)(k0 + 2 * kp) * DDIM + gn + n4;
            bReg[it][0] = *(const float4*)(src);
            bReg[it][1] = *(const float4*)(src + DDIM);
        }
    };
    auto store_tiles = [&](int buf) {
        *(u32x4*)(ldsA[buf] + (size_t)aRow0 * LDSTR + aSeg0 * 8) = aReg[0];
        *(u32x4*)(ldsA[buf] + (size_t)aRow1 * LDSTR + aSeg1 * 8) = aReg[1];
        #pragma unroll
        for (int it = 0; it < 4; ++it) {
            int q  = tid + it * 256;
            int kp = q >> 6;
            int n4 = (q & 63) * 4;
            const float4 w0 = bReg[it][0], w1 = bReg[it][1];
            unsigned int* base0 = (unsigned int*)(ldsB[buf] + (size_t)(n4 + 0) * LDSTR + 2 * kp);
            unsigned int* base1 = (unsigned int*)(ldsB[buf] + (size_t)(n4 + 1) * LDSTR + 2 * kp);
            unsigned int* base2 = (unsigned int*)(ldsB[buf] + (size_t)(n4 + 2) * LDSTR + 2 * kp);
            unsigned int* base3 = (unsigned int*)(ldsB[buf] + (size_t)(n4 + 3) * LDSTR + 2 * kp);
            *base0 = pack2bf(w0.x, w1.x);
            *base1 = pack2bf(w0.y, w1.y);
            *base2 = pack2bf(w0.z, w1.z);
            *base3 = pack2bf(w0.w, w1.w);
        }
    };

    // ---- prologue: stage tile 0 ----
    load_tiles(0);
    store_tiles(0);

    for (int kt = 0; kt < KT; ++kt) {
        const int buf = kt & 1;
        __syncthreads();                    // LDS buf ready / prev reads done

        if (kt + 1 < KT) load_tiles(kt + 1);        // overlap with compute
        if (kt + 2 < KT) {                          // L2 prefetch 2 tiles ahead
            const int k2 = (kt + 2) * KSTEP;
            __builtin_prefetch(Bw + (size_t)(k2 + 2 * (tid >> 6)) * DDIM + gn + (tid & 63) * 4, 0, 1);
            __builtin_prefetch(A + (size_t)(gm + aRow0) * DDIM + k2 + aSeg0 * 8, 0, 1);
        }

        // ---- fragment loads per documented 16-bit A 16x32 layout:
        // lanes 0-15: VGPR0-3 = K 0..7, VGPR4-7 = K 16..23
        // lanes 16-31: VGPR0-3 = K 8..15, VGPR4-7 = K 24..31
        v16bf afrag[4], bfrag[4];
        #pragma unroll
        for (int i = 0; i < 4; ++i) {
            const unsigned short* rA = ldsA[buf] + (size_t)(waveM * 64 + i * 16 + ml) * LDSTR;
            u32x4 lo = *(const u32x4*)(rA + khalf * 8);
            u32x4 hi = *(const u32x4*)(rA + 16 + khalf * 8);
            afrag[i] = make_frag(lo, hi);
        }
        // B 32x16: column = lane&15; lanes 0-15 hold K 0..7,16..23, lanes 16-31 K 8..15,24..31
        #pragma unroll
        for (int j = 0; j < 4; ++j) {
            const unsigned short* cB = ldsB[buf] + (size_t)(waveN * 64 + j * 16 + ml) * LDSTR
                                       + khalf * 8;
            u32x4 lo = *(const u32x4*)(cB);
            u32x4 hi = *(const u32x4*)(cB + 16);
            bfrag[j] = make_frag(lo, hi);
        }

        #pragma unroll
        for (int i = 0; i < 4; ++i)
            #pragma unroll
            for (int j = 0; j < 4; ++j)
                acc[i][j] = __builtin_amdgcn_wmma_f32_16x16x32_bf16(
                    false, afrag[i], false, bfrag[j],
                    (short)0, acc[i][j], false, false);

        if (kt + 1 < KT) store_tiles(buf ^ 1);      // fill the other buffer
    }

    // ---- epilogue ----
    // C/D layout: VGPR v holds M = v (lanes 0-15) / v+8 (lanes 16-31), N = lane&15
    #pragma unroll
    for (int i = 0; i < 4; ++i) {
        const int m0 = gm + waveM * 64 + i * 16 + khalf * 8;
        #pragma unroll
        for (int j = 0; j < 4; ++j) {
            const int col = gn + waveN * 64 + j * 16 + ml;
            const float bia = bias[col];
            if (mode == 1) {
                #pragma unroll
                for (int v = 0; v < 8; ++v) {
                    int row = m0 + v;
                    float val = acc[i][j][v] + bia;
                    val = val > 0.f ? val : 0.f;
                    Hout[(size_t)row * DDIM + col] = f2bf(val);
                }
            } else {
                const float w0 = wcomb[0], w1 = wcomb[1];
                #pragma unroll
                for (int v = 0; v < 8; ++v) {
                    int row = m0 + v;
                    float y = acc[i][j][v] + bia;
                    float g; size_t oidx;
                    if (col < HDIM) {
                        g    = w0 * route_g[(size_t)row * NEXP + expert];
                        oidx = (size_t)row * HDIM + col;
                    } else {
                        g    = w1 * rout_f[(size_t)row * NEXP + expert];
                        oidx = (size_t)NTOK * HDIM + (size_t)row * HDIM + (col - HDIM);
                    }
                    Oout[oidx] += g * y;   // per-expert launches are stream-ordered
                }
            }
        }
    }
}

// ---------------------------------------------------------------------------
// Host-side launcher
// ---------------------------------------------------------------------------
extern "C" void kernel_launch(void* const* d_in, const int* in_sizes, int n_in,
                              void* d_out, int out_size, void* d_ws, size_t ws_size,
                              hipStream_t stream) {
    (void)in_sizes; (void)n_in; (void)out_size; (void)ws_size;

    const float* x    = (const float*)d_in[0];   // [NTOK, DDIM]
    const float* Wg   = (const float*)d_in[1];   // [HDIM, NEXP]
    const float* bg   = (const float*)d_in[2];
    const float* Wf   = (const float*)d_in[3];
    const float* bfv  = (const float*)d_in[4];
    const float* W1   = (const float*)d_in[5];   // [NEXP, DDIM, DDIM]
    const float* b1   = (const float*)d_in[6];   // [NEXP, DDIM]
    const float* W2   = (const float*)d_in[7];
    const float* b2   = (const float*)d_in[8];
    const float* wt   = (const float*)d_in[9];   // [2]
    float*       out  = (float*)d_out;           // out0 | out1, each [NTOK, HDIM]

    char* ws = (char*)d_ws;
    unsigned short* Xbf = (unsigned short*)(ws);                             // 16 MiB
    unsigned short* Hbf = (unsigned short*)(ws + (size_t)NTOK * DDIM * 2);   // 16 MiB
    float*          rg  = (float*)(ws + (size_t)2 * NTOK * DDIM * 2);
    float*          rf  = rg + (size_t)NTOK * NEXP;

    // 1) gating
    gate_kernel<<<NTOK, 256, 0, stream>>>(x, Wg, bg, Wf, bfv, rg, rf);

    // 2) X -> bf16
    {
        int n4 = NTOK * DDIM / 4;                         // 2,097,152
        cvt_bf16_kernel<<<n4 / 256, 256, 0, stream>>>(x, Xbf, n4);
    }

    // 3) zero output accumulators
    {
        int n4 = 2 * NTOK * HDIM / 4;                     // 2,097,152
        zero_kernel<<<n4 / 256, 256, 0, stream>>>(out, n4);
    }

    // 4) per-expert MLP: two bf16-WMMA GEMMs, second fused with gated combine
    dim3 grid(DDIM / BT_N, NTOK / BT_M);                  // (8, 32)
    for (int e = 0; e < NEXP; ++e) {
        const float* W1e = W1 + (size_t)e * DDIM * DDIM;
        const float* W2e = W2 + (size_t)e * DDIM * DDIM;
        const float* b1e = b1 + (size_t)e * DDIM;
        const float* b2e = b2 + (size_t)e * DDIM;
        moe_gemm_kernel<<<grid, 256, 0, stream>>>(Xbf, W1e, b1e, Hbf, out,
                                                  rg, rf, wt, e, /*mode=*/1);
        moe_gemm_kernel<<<grid, 256, 0, stream>>>(Hbf, W2e, b2e, Hbf, out,
                                                  rg, rf, wt, e, /*mode=*/2);
    }
}
